// StandardAttention_34823594836450
// MI455X (gfx1250) — matrix operand order repaired and checked by reading, not compiled, not run on hardware
//
#include <hip/hip_runtime.h>
#include <hip/hip_bf16.h>
#include <math.h>

// ---------------- problem constants ----------------
#define BB 8
#define HH 6
#define NN 1370
#define DD 64
#define CC 384            // HH*DD
#define C3 1152           // 3*CC
#define NP 1408           // N padded to multiple of 64 (22*64)
#define MM (BB*NN)        // 10960 = 685*16 rows
#define MT 685            // M tiles of 16
#define KC 12             // 384/32 k-chunks
#define SCALE 0.125f      // 64^-0.5

typedef __attribute__((ext_vector_type(16))) __bf16 v16bf;
typedef __attribute__((ext_vector_type(8)))  float  v8f;

// ---------------- workspace layout (bytes) ----------------
#define SZ_XB    ((size_t)MM*CC)            // ushort counts
#define SZ_WQKV  ((size_t)C3*CC)
#define SZ_WPROJ ((size_t)CC*CC)
#define SZ_QKV   ((size_t)BB*HH*NP*DD)      // each of Qb/Kb/Vt
#define SZ_OB    ((size_t)MM*CC)

#define OFF_XB    ((size_t)0)
#define OFF_WQKV  (OFF_XB   + SZ_XB*2)
#define OFF_WPROJ (OFF_WQKV + SZ_WQKV*2)
#define OFF_QB    (OFF_WPROJ+ SZ_WPROJ*2)
#define OFF_KB    (OFF_QB   + SZ_QKV*2)
#define OFF_VT    (OFF_KB   + SZ_QKV*2)
#define OFF_OB    (OFF_VT   + SZ_QKV*2)

__device__ __forceinline__ unsigned short f2bf(float x) {
    unsigned int u = __float_as_uint(x);
    unsigned int r = u + 0x7FFFu + ((u >> 16) & 1u);
    return (unsigned short)(r >> 16);
}

__device__ __forceinline__ v8f wmma_bf16(v16bf a, v16bf b, v8f c) {
    return __builtin_amdgcn_wmma_f32_16x16x32_bf16(
        false, a, false, b, (short)0, c, false, false);
}

// A-matrix fragment: lane = row (lane%16), half = lane/16.
// VGPR i holds k = kbase + 16*(i/4) + 8*half + 2*(i%4) (+1)
__device__ __forceinline__ v16bf load_frag_A(const unsigned short* base,
                                             int row_stride, int row,
                                             int kbase, int hf) {
    union { unsigned int u[8]; v16bf v; } f;
#pragma unroll
    for (int i = 0; i < 8; ++i) {
        int k = kbase + ((i >> 2) << 4) + (hf << 3) + ((i & 3) << 1);
        f.u[i] = *(const unsigned int*)(base + (size_t)row * row_stride + k);
    }
    return f.v;
}

// B-matrix fragment from B^T storage (rows of W for X@W^T):
// lane = col (lane%16); VGPR j holds k = kbase + 16*(j/4) + 8*half + 2*(j%4) (+1)
__device__ __forceinline__ v16bf load_frag_BT(const unsigned short* base,
                                              int row_stride, int colrow,
                                              int kbase, int hf) {
    union { unsigned int u[8]; v16bf v; } f;
#pragma unroll
    for (int j = 0; j < 8; ++j) {
        int k = kbase + ((j >> 2) << 4) + (hf << 3) + ((j & 3) << 1);
        f.u[j] = *(const unsigned int*)(base + (size_t)colrow * row_stride + k);
    }
    return f.v;
}

// ---------------- kernel 0: f32 -> bf16 convert ----------------
__global__ void k_cvt_bf16(const float* __restrict__ in,
                           unsigned short* __restrict__ out, int n) {
    int i = blockIdx.x * blockDim.x + threadIdx.x;
    if (i < n) out[i] = f2bf(in[i]);
}

// ---------------- kernel 1: zero (uint4) ----------------
__global__ void k_zero16(uint4* __restrict__ p, int n16) {
    int i = blockIdx.x * blockDim.x + threadIdx.x;
    if (i < n16) p[i] = make_uint4(0u, 0u, 0u, 0u);
}

// ---------------- kernel 2: QKV GEMM + scatter epilogue ----------------
// out tiles: grid.x covers M tiles (4 waves/block), grid.y = 72 col tiles
__global__ void k_qkv_gemm(const unsigned short* __restrict__ xb,
                           const unsigned short* __restrict__ wq,
                           const float* __restrict__ qkv_b,
                           unsigned short* __restrict__ Qb,
                           unsigned short* __restrict__ Kb,
                           unsigned short* __restrict__ Vt) {
    int wave = threadIdx.x >> 5, lane = threadIdx.x & 31;
    int mtile = blockIdx.x * 4 + wave;
    if (mtile >= MT) return;
    int ot = blockIdx.y;
    int hf = lane >> 4, col = lane & 15;

    v8f acc = {};
#pragma unroll
    for (int kc = 0; kc < KC; ++kc) {
        v16bf a = load_frag_A (xb, CC, mtile * 16 + col, kc * 32, hf);
        v16bf b = load_frag_BT(wq, CC, ot    * 16 + col, kc * 32, hf);
        acc = wmma_bf16(a, b, acc);
    }

    int o = ot * 16 + col;            // 0..1151
    float bias  = qkv_b[o];
    int which   = o / CC;             // 0=q 1=k 2=v
    int rem     = o - which * CC;
    int h       = rem >> 6;
    int d       = rem & 63;
#pragma unroll
    for (int r = 0; r < 8; ++r) {
        int gM = mtile * 16 + hf * 8 + r;
        int b_ = gM / NN;
        int n  = gM - b_ * NN;
        float val = acc[r] + bias;
        size_t bh = (size_t)(b_ * HH + h);
        if (which == 0) {
            Qb[(bh * NP + n) * DD + d] = f2bf(val * SCALE);
        } else if (which == 1) {
            Kb[(bh * NP + n) * DD + d] = f2bf(val);
        } else {
            Vt[(bh * DD + d) * NP + n] = f2bf(val);
        }
    }
}

// ---------------- kernel 3: flash attention ----------------
// grid = (B*H, NP/64); block = 128 (4 waves); each wave: 16 query rows
__global__ void k_attn(const unsigned short* __restrict__ Qb,
                       const unsigned short* __restrict__ Kb,
                       const unsigned short* __restrict__ Vt,
                       unsigned short* __restrict__ Ob) {
    __shared__ unsigned short ldsP[4][16 * 32];
    int wave = threadIdx.x >> 5, lane = threadIdx.x & 31;
    int bh   = blockIdx.x;
    int q0   = blockIdx.y * 64 + wave * 16;
    int hf = lane >> 4, col = lane & 15;

    const unsigned short* Qp = Qb + (size_t)bh * NP * DD;
    const unsigned short* Kp = Kb + (size_t)bh * NP * DD;
    const unsigned short* Vp = Vt + (size_t)bh * DD * NP;
    unsigned short* myP = &ldsP[wave][0];

    // Q fragments (16 rows x 64 features, as two K=32 A-frags); scale pre-baked
    v16bf qa0 = load_frag_A(Qp, DD, q0 + col, 0,  hf);
    v16bf qa1 = load_frag_A(Qp, DD, q0 + col, 32, hf);

    float m_i[8], l_i[8];
    v8f oacc[4];
#pragma unroll
    for (int r = 0; r < 8; ++r) { m_i[r] = -3.0e38f; l_i[r] = 0.f; }
#pragma unroll
    for (int dt = 0; dt < 4; ++dt) oacc[dt] = (v8f){};

    for (int m0 = 0; m0 < NP; m0 += 32) {
        if (m0 + 32 < NP) {
            __builtin_prefetch(Kp + (size_t)(m0 + 32) * DD, 0, 1);
            __builtin_prefetch(Vp + (size_t)col * NP + m0 + 32, 0, 1);
        }
        // ---- S = Q K^T for 32 keys (two 16-col subtiles) ----
        v16bf kb;
        v8f s0 = {}, s1 = {};
        kb = load_frag_BT(Kp, DD, m0 + col,      0,  hf); s0 = wmma_bf16(qa0, kb, s0);
        kb = load_frag_BT(Kp, DD, m0 + col,      32, hf); s0 = wmma_bf16(qa1, kb, s0);
        kb = load_frag_BT(Kp, DD, m0 + 16 + col, 0,  hf); s1 = wmma_bf16(qa0, kb, s1);
        kb = load_frag_BT(Kp, DD, m0 + 16 + col, 32, hf); s1 = wmma_bf16(qa1, kb, s1);

        bool v0 = (m0 + col)      < NN;
        bool v1 = (m0 + 16 + col) < NN;
        float p0[8], p1[8], rmax[8], rsum[8];
#pragma unroll
        for (int r = 0; r < 8; ++r) {
            float a = v0 ? s0[r] : -3.0e38f;
            float b = v1 ? s1[r] : -3.0e38f;
            p0[r] = a; p1[r] = b;
            rmax[r] = fmaxf(a, b);
        }
        // row-wise max across the 16 lanes holding this row's columns
#pragma unroll
        for (int r = 0; r < 8; ++r) {
#pragma unroll
            for (int msk = 1; msk < 16; msk <<= 1)
                rmax[r] = fmaxf(rmax[r], __shfl_xor(rmax[r], msk));
        }
#pragma unroll
        for (int r = 0; r < 8; ++r) {
            float mnew  = fmaxf(m_i[r], rmax[r]);
            float alpha = __expf(m_i[r] - mnew);
            p0[r] = __expf(p0[r] - mnew);
            p1[r] = __expf(p1[r] - mnew);
            rsum[r] = p0[r] + p1[r];
            m_i[r] = mnew;
            l_i[r] = l_i[r] * alpha;
#pragma unroll
            for (int dt = 0; dt < 4; ++dt) oacc[dt][r] *= alpha;
        }
#pragma unroll
        for (int r = 0; r < 8; ++r) {
#pragma unroll
            for (int msk = 1; msk < 16; msk <<= 1)
                rsum[r] += __shfl_xor(rsum[r], msk);
            l_i[r] += rsum[r];
        }
        // ---- P (C-layout) -> LDS -> A-layout fragment ----
#pragma unroll
        for (int r = 0; r < 8; ++r) {
            int row = hf * 8 + r;
            myP[row * 32 + col]      = f2bf(p0[r]);
            myP[row * 32 + 16 + col] = f2bf(p1[r]);
        }
        v16bf pa = load_frag_A(myP, 32, col, 0, hf);
        // ---- O += P V ----
#pragma unroll
        for (int dt = 0; dt < 4; ++dt) {
            v16bf vb = load_frag_BT(Vp, NP, dt * 16 + col, m0, hf);
            oacc[dt] = wmma_bf16(pa, vb, oacc[dt]);
        }
    }

    // ---- finalize: O /= l, store bf16 into [B,N,C] ----
    int b_ = bh / HH, h = bh - b_ * HH;
#pragma unroll
    for (int r = 0; r < 8; ++r) {
        int n = q0 + hf * 8 + r;
        if (n >= NN) continue;
        float inv = 1.0f / l_i[r];
        size_t rowbase = ((size_t)b_ * NN + n) * CC + h * 64;
#pragma unroll
        for (int dt = 0; dt < 4; ++dt)
            Ob[rowbase + dt * 16 + col] = f2bf(oacc[dt][r] * inv);
    }
}

// ---------------- kernel 4: output projection ----------------
__global__ void k_proj_gemm(const unsigned short* __restrict__ Ob,
                            const unsigned short* __restrict__ wp,
                            const float* __restrict__ proj_b,
                            float* __restrict__ out) {
    int wave = threadIdx.x >> 5, lane = threadIdx.x & 31;
    int mtile = blockIdx.x * 4 + wave;
    if (mtile >= MT) return;
    int ot = blockIdx.y;
    int hf = lane >> 4, col = lane & 15;

    v8f acc = {};
#pragma unroll
    for (int kc = 0; kc < KC; ++kc) {
        v16bf a = load_frag_A (Ob, CC, mtile * 16 + col, kc * 32, hf);
        v16bf b = load_frag_BT(wp, CC, ot    * 16 + col, kc * 32, hf);
        acc = wmma_bf16(a, b, acc);
    }
    int o = ot * 16 + col;
    float bias = proj_b[o];
#pragma unroll
    for (int r = 0; r < 8; ++r) {
        int gM = mtile * 16 + hf * 8 + r;
        out[(size_t)gM * CC + o] = acc[r] + bias;
    }
}

// ---------------- host launcher ----------------
extern "C" void kernel_launch(void* const* d_in, const int* in_sizes, int n_in,
                              void* d_out, int out_size, void* d_ws, size_t ws_size,
                              hipStream_t stream) {
    const float* x      = (const float*)d_in[0];
    const float* qkv_w  = (const float*)d_in[1];
    const float* qkv_b  = (const float*)d_in[2];
    const float* proj_w = (const float*)d_in[3];
    const float* proj_b = (const float*)d_in[4];
    float* out = (float*)d_out;

    char* ws = (char*)d_ws;
    unsigned short* xb  = (unsigned short*)(ws + OFF_XB);
    unsigned short* wq  = (unsigned short*)(ws + OFF_WQKV);
    unsigned short* wp  = (unsigned short*)(ws + OFF_WPROJ);
    unsigned short* Qb  = (unsigned short*)(ws + OFF_QB);
    unsigned short* Kb  = (unsigned short*)(ws + OFF_KB);
    unsigned short* Vt  = (unsigned short*)(ws + OFF_VT);
    unsigned short* Ob  = (unsigned short*)(ws + OFF_OB);

    // 1) converts
    {
        int n = (int)SZ_XB;
        k_cvt_bf16<<<(n + 255) / 256, 256, 0, stream>>>(x, xb, n);
        n = (int)SZ_WQKV;
        k_cvt_bf16<<<(n + 255) / 256, 256, 0, stream>>>(qkv_w, wq, n);
        n = (int)SZ_WPROJ;
        k_cvt_bf16<<<(n + 255) / 256, 256, 0, stream>>>(proj_w, wp, n);
    }
    // 2) zero Q/K/Vt (padding rows/cols must be 0)
    {
        int n16 = (int)((SZ_QKV * 2 * 3) / 16);
        k_zero16<<<(n16 + 255) / 256, 256, 0, stream>>>((uint4*)(ws + OFF_QB), n16);
    }
    // 3) QKV projection
    {
        dim3 grid((MT + 3) / 4, C3 / 16);
        k_qkv_gemm<<<grid, 128, 0, stream>>>(xb, wq, qkv_b, Qb, Kb, Vt);
    }
    // 4) flash attention
    {
        dim3 grid(BB * HH, NP / 64);
        k_attn<<<grid, 128, 0, stream>>>(Qb, Kb, Vt, Ob);
    }
    // 5) output projection
    {
        dim3 grid((MT + 3) / 4, CC / 16);
        k_proj_gemm<<<grid, 128, 0, stream>>>(Ob, wp, proj_b, out);
    }
}
